// MambaResidualBlock_18150531793387
// MI455X (gfx1250) — matrix operand order repaired and checked by reading, not compiled, not run on hardware
//
#include <hip/hip_runtime.h>

// ---------------------------------------------------------------------------
// MambaResidualBlock for MI455X (gfx1250, wave32, WMMA).
// All four GEMMs run on v_wmma_f32_16x16x32_f16 (f32 accumulate).
// Working set (~100MB) fits in the 192MB L2, so GEMMs are WMMA-bound.
// Round 5: fix async builtin arg types -- it takes (global v4i*, lds v4i*,
// imm offset, imm cpol) per the clang diagnostic.
// ---------------------------------------------------------------------------

typedef __attribute__((ext_vector_type(16))) _Float16 v16h;
typedef __attribute__((ext_vector_type(8)))  _Float16 v8h;
typedef __attribute__((ext_vector_type(8)))  float    v8f;
typedef __attribute__((ext_vector_type(4)))  int      v4i;

#define GLOBAL_AS __attribute__((address_space(1)))
#define LDS_AS    __attribute__((address_space(3)))

#if defined(__gfx1250__) && \
    __has_builtin(__builtin_amdgcn_global_load_async_to_lds_b128) && \
    __has_builtin(__builtin_amdgcn_s_wait_asynccnt)
#define USE_ASYNC_LDS 1
#else
#define USE_ASYNC_LDS 0
#endif

static constexpr int Bc     = 2;
static constexpr int Lc     = 2048;
static constexpr int DIMc   = 1024;
static constexpr int Sc     = 16;
static constexpr int INNERc = 1024;
static constexpr int TOK    = Bc * Lc;         // 4096 tokens
static constexpr int TWO_INNER = 2 * INNERc;   // 2048

// ---- workspace layout (bytes, all 256-aligned) ----------------------------
static constexpr size_t OFF_H    = 0;                                // h f16 (4096x1024)
static constexpr size_t OFF_WIN  = OFF_H    + (size_t)TOK*DIMc*2;    // W_in f16 (2048x1024)
static constexpr size_t OFF_WDT  = OFF_WIN  + (size_t)TWO_INNER*DIMc*2;
static constexpr size_t OFF_WX   = OFF_WDT  + (size_t)INNERc*DIMc*2;
static constexpr size_t OFF_WOUT = OFF_WX   + (size_t)2*Sc*INNERc*2;
static constexpr size_t OFF_XZ   = OFF_WOUT + (size_t)DIMc*INNERc*2; // xz f32 (4096x2048)
static constexpr size_t OFF_XCH  = OFF_XZ   + (size_t)TOK*TWO_INNER*4;
static constexpr size_t OFF_XCF  = OFF_XCH  + (size_t)TOK*INNERc*2;
static constexpr size_t OFF_DT   = OFF_XCF  + (size_t)TOK*INNERc*4;
static constexpr size_t OFF_BC   = OFF_DT   + (size_t)TOK*INNERc*4;
static constexpr size_t OFF_YH   = OFF_H;                            // reuse h for y (f16)

// ---------------------------------------------------------------------------
// f32 -> f16 convert
__global__ void __launch_bounds__(256)
cvt_f32_f16(const float* __restrict__ src, _Float16* __restrict__ dst, int n) {
    int i = blockIdx.x * 256 + threadIdx.x;
    if (i < n) dst[i] = (_Float16)src[i];
}

// ---------------------------------------------------------------------------
// LayerNorm over DIM, emit f16 rows for WMMA A-operand. One block per token.
__global__ void __launch_bounds__(256)
layernorm_f16(const float* __restrict__ x, const float* __restrict__ gamma,
              const float* __restrict__ beta, _Float16* __restrict__ h) {
    const int row = blockIdx.x;
    const float* xr = x + (size_t)row * DIMc;
    float s = 0.f, s2 = 0.f;
    for (int i = threadIdx.x; i < DIMc; i += 256) {
        float v = xr[i]; s += v; s2 += v * v;
    }
    #pragma unroll
    for (int off = 16; off > 0; off >>= 1) {
        s  += __shfl_down(s,  off);
        s2 += __shfl_down(s2, off);
    }
    __shared__ float ss[8], ss2[8];
    if ((threadIdx.x & 31) == 0) { ss[threadIdx.x >> 5] = s; ss2[threadIdx.x >> 5] = s2; }
    __syncthreads();
    float ts = 0.f, ts2 = 0.f;
    #pragma unroll
    for (int i = 0; i < 8; ++i) { ts += ss[i]; ts2 += ss2[i]; }
    const float mu   = ts / DIMc;
    const float var  = ts2 / DIMc - mu * mu;
    const float rstd = rsqrtf(var + 1e-5f);
    for (int i = threadIdx.x; i < DIMc; i += 256) {
        float v = (xr[i] - mu) * rstd * gamma[i] + beta[i];
        h[(size_t)row * DIMc + i] = (_Float16)v;
    }
}

// ---------------------------------------------------------------------------
// Fragment load: two global b128 loads at p and p+gap halves, forced AS(1).
__device__ __forceinline__ v16h load16(const _Float16* p, int gap) {
    v8h lo = *(const GLOBAL_AS v8h*)(p);
    v8h hi = *(const GLOBAL_AS v8h*)(p + gap);
    v16h f;
    #pragma unroll
    for (int i = 0; i < 8; ++i) { f[i] = lo[i]; f[8 + i] = hi[i]; }
    return f;
}

// Fragment load from LDS: two ds_load_b128 at p and p+32 bytes.
__device__ __forceinline__ v16h load16_lds(const char* p) {
    v8h lo = *(const v8h*)(p);
    v8h hi = *(const v8h*)(p + 32);
    v16h f;
    #pragma unroll
    for (int i = 0; i < 8; ++i) { f[i] = lo[i]; f[8 + i] = hi[i]; }
    return f;
}

__device__ __forceinline__ v8f wmma_f16(v16h a, v16h b, v8f c) {
    return __builtin_amdgcn_wmma_f32_16x16x32_f16(
        false, a, false, b, (short)0, c, false, false);
}

// ---------------------------------------------------------------------------
// Staged GEMM: C[M,N] = A[M,K f16] * W[N,K f16]^T (+bias[N]) (+add[M,N]).
// Fixed tiling: wave tile 64x64 (MT=NT=4), block 2x4 waves -> 128x256 tile.
// A is staged per-block through LDS with double-buffered async copies over
// CK=64-wide K chunks; B is loaded direct-from-global (L2-resident weights).
// M % 128 == 0, N % 256 == 0, K % 64 == 0. Branch-free around all WMMA.
static constexpr int CK      = 64;               // K halves per stage
static constexpr int ROWPAD  = 144;              // LDS bytes/row: 128 data + 16 pad
static constexpr int SLAB    = 128 * ROWPAD;     // 18432 B per buffer

template<bool HAS_BIAS, bool HAS_ADD>
__global__ void __launch_bounds__(256)
gemm_f16_wmma_stage(const _Float16* __restrict__ A, const _Float16* __restrict__ W,
                    const float* __restrict__ bias, const float* __restrict__ addsrc,
                    float* __restrict__ C, int N, int K) {
    __shared__ char smem[2 * SLAB];
    const int tid    = threadIdx.x;
    const int lane   = tid & 31;
    const int wave   = tid >> 5;
    const int mBlock = blockIdx.x * 128;
    const int mWaveL = (wave >> 2) * 64;                     // wave-local row base
    const int nWave  = blockIdx.y * 256 + (wave & 3) * 64;
    const int lm     = lane & 15;

    // B fragment pointers (direct global).
    const _Float16* pB[4];
    #pragma unroll
    for (int j = 0; j < 4; ++j)
        pB[j] = W + (size_t)(nWave + j * 16 + lm) * K + ((lane >> 4) << 3);

    v8f acc[4][4];
    #pragma unroll
    for (int i = 0; i < 4; ++i)
        #pragma unroll
        for (int j = 0; j < 4; ++j)
            #pragma unroll
            for (int r = 0; r < 8; ++r) acc[i][j][r] = 0.f;

    // Stage issue: copy A[mBlock..+127, kc..kc+63] into smem buffer `buf`.
    // 1024 x 16B transfers; 4 per thread. row = q>>3, 16B-seg = q&7.
    auto issue_stage = [&](int kc, int buf) {
        char* dst = smem + buf * SLAB;
        #pragma unroll
        for (int c = 0; c < 4; ++c) {
            const int q   = tid * 4 + c;
            const int row = q >> 3;
            const int seg = q & 7;
            const _Float16* g = A + (size_t)(mBlock + row) * K + kc + seg * 8;
            char* l = dst + row * ROWPAD + seg * 16;
#if USE_ASYNC_LDS
            __builtin_amdgcn_global_load_async_to_lds_b128(
                (GLOBAL_AS v4i*)g, (LDS_AS v4i*)l, 0, 0);
#else
            *(v8h*)l = *(const GLOBAL_AS v8h*)g;
#endif
        }
    };

    issue_stage(0, 0);
    int buf = 0;
    for (int kc = 0; kc < K; kc += CK) {
        const bool hasNext = (kc + CK) < K;
        if (hasNext) issue_stage(kc + CK, buf ^ 1);          // prefetch next chunk
#if USE_ASYNC_LDS
        if (hasNext) __builtin_amdgcn_s_wait_asynccnt(4);    // chunk kc done (in order)
        else         __builtin_amdgcn_s_wait_asynccnt(0);
#endif
        __syncthreads();                                     // chunk kc visible to all
        const char* sl = smem + buf * SLAB;
        #pragma unroll
        for (int ks = 0; ks < 2; ++ks) {                     // two 32-wide K-steps
            v16h a[4];
            #pragma unroll
            for (int i = 0; i < 4; ++i) {
                const char* p = sl + (mWaveL + i * 16 + lm) * ROWPAD
                                   + ks * 64 + ((lane >> 4) << 4);
                a[i] = load16_lds(p);
            }
            #pragma unroll
            for (int j = 0; j < 4; ++j) {
                const v16h b = load16(pB[j], 16);
                pB[j] += 32;
                #pragma unroll
                for (int i = 0; i < 4; ++i)
                    acc[i][j] = wmma_f16(a[i], b, acc[i][j]);
            }
        }
        __syncthreads();                                     // reads done before overwrite
        buf ^= 1;
    }

    // C/D layout: VGPR r -> row mTile + 8*(lane>>4) + r, col nTile + (lane&15).
    const int mWave  = mBlock + mWaveL;
    const int colOfs = lane & 15;
    const int rowOfs = (lane >> 4) << 3;
    #pragma unroll
    for (int i = 0; i < 4; ++i) {
        const int mTile = mWave + i * 16;
        #pragma unroll
        for (int j = 0; j < 4; ++j) {
            const int col = nWave + j * 16 + colOfs;
            float bv = 0.f;
            if constexpr (HAS_BIAS) bv = bias[col];
            #pragma unroll
            for (int r = 0; r < 8; ++r) {
                const int row = mTile + rowOfs + r;
                float v = acc[i][j][r] + bv;
                if constexpr (HAS_ADD)
                    v += *(const GLOBAL_AS float*)(addsrc + (size_t)row * N + col);
                *(GLOBAL_AS float*)(C + (size_t)row * N + col) = v;
            }
        }
    }
}

// ---------------------------------------------------------------------------
// Direct-global GEMM (used for the tiny N=32 BC projection).
// Wave tile (MT*16)x(NT*16); block WM x WN waves (WM*WN == 8).
template<int MT, int NT, int WM, int WN>
__global__ void __launch_bounds__(256)
gemm_f16_wmma(const _Float16* __restrict__ A, const _Float16* __restrict__ W,
              float* __restrict__ C, int N, int K) {
    const int lane  = threadIdx.x & 31;
    const int wave  = threadIdx.x >> 5;
    const int mWave = blockIdx.x * (WM * MT * 16) + (wave / WN) * (MT * 16);
    const int nWave = blockIdx.y * (WN * NT * 16) + (wave % WN) * (NT * 16);
    const int lm = lane & 15;

    const _Float16* pA[MT];
    #pragma unroll
    for (int i = 0; i < MT; ++i)
        pA[i] = A + (size_t)(mWave + i * 16 + lm) * K + ((lane >> 4) << 3);
    const _Float16* pB[NT];
    #pragma unroll
    for (int j = 0; j < NT; ++j)
        pB[j] = W + (size_t)(nWave + j * 16 + lm) * K + ((lane >> 4) << 3);

    v8f acc[MT][NT];
    #pragma unroll
    for (int i = 0; i < MT; ++i)
        #pragma unroll
        for (int j = 0; j < NT; ++j)
            #pragma unroll
            for (int r = 0; r < 8; ++r) acc[i][j][r] = 0.f;

    for (int kk = 0; kk < K; kk += 32) {
        v16h a[MT];
        #pragma unroll
        for (int i = 0; i < MT; ++i) { a[i] = load16(pA[i], 16); pA[i] += 32; }
        #pragma unroll
        for (int j = 0; j < NT; ++j) {
            const v16h b = load16(pB[j], 16);
            pB[j] += 32;
            #pragma unroll
            for (int i = 0; i < MT; ++i)
                acc[i][j] = wmma_f16(a[i], b, acc[i][j]);
        }
    }

    const int colOfs = lane & 15;
    const int rowOfs = (lane >> 4) << 3;
    #pragma unroll
    for (int i = 0; i < MT; ++i) {
        const int mTile = mWave + i * 16;
        #pragma unroll
        for (int j = 0; j < NT; ++j) {
            const int col = nWave + j * 16 + colOfs;
            #pragma unroll
            for (int r = 0; r < 8; ++r) {
                const int row = mTile + rowOfs + r;
                *(GLOBAL_AS float*)(C + (size_t)row * N + col) = acc[i][j][r];
            }
        }
    }
}

// ---------------------------------------------------------------------------
// Depthwise conv (K=3, same-pad) over L + SiLU. xx = xz[..., :INNER].
__global__ void __launch_bounds__(256)
conv_silu(const float* __restrict__ xz, const float* __restrict__ conv_w,
          _Float16* __restrict__ xc_h, float* __restrict__ xc_f) {
    const int idx = blockIdx.x * 256 + threadIdx.x;     // over TOK*INNER
    if (idx >= TOK * INNERc) return;
    const int d = idx & (INNERc - 1);
    const int t = idx >> 10;                            // global token
    const int l = t & (Lc - 1);
    const float w0 = conv_w[d * 3 + 0], w1 = conv_w[d * 3 + 1], w2 = conv_w[d * 3 + 2];
    const float xm = (l > 0)      ? xz[(size_t)(t - 1) * TWO_INNER + d] : 0.f;
    const float x0 =                xz[(size_t)t       * TWO_INNER + d];
    const float xp = (l < Lc - 1) ? xz[(size_t)(t + 1) * TWO_INNER + d] : 0.f;
    float v = xm * w0 + x0 * w1 + xp * w2;
    v = v / (1.f + __expf(-v));                         // SiLU
    xc_h[idx] = (_Float16)v;
    xc_f[idx] = v;
}

// ---------------------------------------------------------------------------
// Selective scan: one lane per (b,d,s); 16-lane shuffle reduce over s.
// Fuses softplus(dt), dA=exp(dt*A), state update, C-contraction, +D*xc, *silu(z).
__global__ void __launch_bounds__(256)
scan_fused(const float* __restrict__ dt_raw, const float* __restrict__ BCbuf,
           const float* __restrict__ A_log, const float* __restrict__ D_par,
           const float* __restrict__ xc_f, const float* __restrict__ xz,
           _Float16* __restrict__ y_h) {
    const int tid = blockIdx.x * 256 + threadIdx.x;     // 0 .. B*INNER*S-1 = 32768
    const int s = tid & (Sc - 1);
    const int d = (tid >> 4) & (INNERc - 1);
    const int b = tid >> 14;                            // (tid>>4)/1024
    const float Av = -__expf(A_log[d * Sc + s]);
    const float Dp = D_par[d];
    float state = 0.f;
    const size_t tokBase = (size_t)b * Lc;
    for (int l = 0; l < Lc; ++l) {
        const size_t t = tokBase + l;
        float dtv = dt_raw[t * INNERc + d];
        dtv = (dtv > 20.f) ? dtv : log1pf(__expf(dtv));      // softplus
        const float Bv = BCbuf[t * (2 * Sc) + s];
        const float Cv = BCbuf[t * (2 * Sc) + Sc + s];
        state = state * __expf(dtv * Av) + dtv * Bv;
        float contrib = state * Cv;
        contrib += __shfl_xor(contrib, 1, 16);
        contrib += __shfl_xor(contrib, 2, 16);
        contrib += __shfl_xor(contrib, 4, 16);
        contrib += __shfl_xor(contrib, 8, 16);
        if (s == 0) {
            const float y = contrib + Dp * xc_f[t * INNERc + d];
            const float z = xz[t * TWO_INNER + INNERc + d];
            const float sz = z / (1.f + __expf(-z));          // silu(z)
            y_h[t * INNERc + d] = (_Float16)(y * sz);
        }
    }
}

// ---------------------------------------------------------------------------
extern "C" void kernel_launch(void* const* d_in, const int* in_sizes, int n_in,
                              void* d_out, int out_size, void* d_ws, size_t ws_size,
                              hipStream_t stream) {
    const float* x      = (const float*)d_in[0];
    const float* gamma  = (const float*)d_in[1];
    const float* beta   = (const float*)d_in[2];
    const float* W_in   = (const float*)d_in[3];
    const float* conv_w = (const float*)d_in[4];
    const float* W_x    = (const float*)d_in[5];
    const float* W_dt   = (const float*)d_in[6];
    const float* b_dt   = (const float*)d_in[7];
    const float* A_log  = (const float*)d_in[8];
    const float* D_par  = (const float*)d_in[9];
    const float* W_out  = (const float*)d_in[10];
    float* out = (float*)d_out;

    char* ws = (char*)d_ws;
    _Float16* h_f16    = (_Float16*)(ws + OFF_H);
    _Float16* Win_f16  = (_Float16*)(ws + OFF_WIN);
    _Float16* Wdt_f16  = (_Float16*)(ws + OFF_WDT);
    _Float16* Wx_f16   = (_Float16*)(ws + OFF_WX);
    _Float16* Wout_f16 = (_Float16*)(ws + OFF_WOUT);
    float*    xz       = (float*)(ws + OFF_XZ);
    _Float16* xc_h     = (_Float16*)(ws + OFF_XCH);
    float*    xc_f     = (float*)(ws + OFF_XCF);
    float*    dt_f     = (float*)(ws + OFF_DT);
    float*    BCbuf    = (float*)(ws + OFF_BC);
    _Float16* y_h      = (_Float16*)(ws + OFF_YH);   // aliases h (dead after GEMM1)

    const dim3 blk(256);

    // weights -> f16
    cvt_f32_f16<<<dim3((TWO_INNER*DIMc + 255)/256), blk, 0, stream>>>(W_in,  Win_f16,  TWO_INNER*DIMc);
    cvt_f32_f16<<<dim3((INNERc*DIMc    + 255)/256), blk, 0, stream>>>(W_dt,  Wdt_f16,  INNERc*DIMc);
    cvt_f32_f16<<<dim3((2*Sc*INNERc    + 255)/256), blk, 0, stream>>>(W_x,   Wx_f16,   2*Sc*INNERc);
    cvt_f32_f16<<<dim3((DIMc*INNERc    + 255)/256), blk, 0, stream>>>(W_out, Wout_f16, DIMc*INNERc);

    // layernorm -> h (f16)
    layernorm_f16<<<dim3(TOK), blk, 0, stream>>>(x, gamma, beta, h_f16);

    // xz = h @ W_in^T  (4096 x 2048): staged, block tile 128x256
    gemm_f16_wmma_stage<false, false>
        <<<dim3(TOK/128, TWO_INNER/256), blk, 0, stream>>>(
            h_f16, Win_f16, nullptr, nullptr, xz, TWO_INNER, DIMc);

    // depthwise conv + SiLU -> xc (f16 + f32)
    conv_silu<<<dim3((TOK*INNERc + 255)/256), blk, 0, stream>>>(xz, conv_w, xc_h, xc_f);

    // dt = xc @ W_dt^T + b_dt  (4096 x 1024): staged
    gemm_f16_wmma_stage<true, false>
        <<<dim3(TOK/128, INNERc/256), blk, 0, stream>>>(
            xc_h, Wdt_f16, b_dt, nullptr, dt_f, INNERc, INNERc);

    // BC = xc @ W_x^T  (4096 x 32): direct, block tile 256x32 (exact)
    gemm_f16_wmma<2, 2, 8, 1>
        <<<dim3(TOK/256, 1), blk, 0, stream>>>(
            xc_h, Wx_f16, BCbuf, 2*Sc, INNERc);

    // selective scan + gating -> y (f16)
    scan_fused<<<dim3(Bc*INNERc*Sc/256), blk, 0, stream>>>(
        dt_f, BCbuf, A_log, D_par, xc_f, xz, y_h);

    // out = y @ W_out^T + residual  (4096 x 1024): staged
    gemm_f16_wmma_stage<false, true>
        <<<dim3(TOK/128, INNERc/256), blk, 0, stream>>>(
            y_h, Wout_f16, nullptr, x, out, INNERc, INNERc);
}
